// BasicTransformerBlock_37812892074003
// MI455X (gfx1250) — compile-verified, matrix operand order checked by reading, not run on hardware
//
#include <hip/hip_runtime.h>
#include <math.h>

// ---------------------------------------------------------------- types
typedef __attribute__((ext_vector_type(16))) _Float16 v16h;
typedef __attribute__((ext_vector_type(8)))  _Float16 v8h;
typedef __attribute__((ext_vector_type(8)))  float    v8f;

#define NH  16
#define DH  64
#define DAc 1024
#define DMc 1024
#define DCc 768
#define DFFc 4096
#define Bc  8
#define Sc  1024
#define Tcc 256

// ---------------------------------------------------------------- WMMA helpers
__device__ __forceinline__ v8f wmma_f16(v16h a, v16h b, v8f c) {
  return __builtin_amdgcn_wmma_f32_16x16x32_f16(false, a, false, b, (short)0, c,
                                                false, false);
}

__device__ __forceinline__ v16h combine(v8h lo, v8h hi) {
  v16h r;
#pragma unroll
  for (int i = 0; i < 8; ++i) { r[i] = lo[i]; r[i + 8] = hi[i]; }
  return r;
}

// 16x32 f16 fragment (A layout; dense B layout identical) from row-major
// storage with COMPILE-TIME leading dimension LD: all fragment addresses
// become one per-stream base + constant byte deltas (folded into the 24-bit
// instruction offset of global_load_b128).
// lanes 0-15: row = lane, K chunks {0..7, 16..23}
// lanes 16-31: row = lane-16, K chunks {8..15, 24..31}
template <int LD>
__device__ __forceinline__ v16h ldgc(const _Float16* __restrict__ base, int off) {
  int lane = (int)(threadIdx.x & 31);
  int o = off + (lane & 15) * LD + ((lane >> 4) << 3);
  v8h lo = *(const v8h*)(base + o);
  v8h hi = *(const v8h*)(base + o + 16);
  return combine(lo, hi);
}

// Same fragment layout from an LDS tile (pointer form, constant ld=32).
__device__ __forceinline__ v16h ldlds(const _Float16* p) {
  int lane = (int)(threadIdx.x & 31);
  const _Float16* q = p + (lane & 15) * 32 + ((lane >> 4) << 3);
  v8h lo = *(const v8h*)(q);
  v8h hi = *(const v8h*)(q + 16);
  return combine(lo, hi);
}

// ---------------------------------------------------------------- generic GEMM
// WG tile 64x256 (8 waves as 2Mx4N); wave tile 32x64 = 8 accum fragments.
// A: MxK row-major, WT: NxK row-major (both stride K, compile-time).
// c[i*4+j] covers rows m0+16*i, cols n0+16*j.
template <int K>
__device__ __forceinline__ void gemm_core(const _Float16* __restrict__ A,
                                          const _Float16* __restrict__ WT,
                                          int m0, int n0, v8f c[8]) {
  int ao0 = m0 * K;
  int bo0 = n0 * K;
  for (int k0 = 0; k0 < K; k0 += 32) {
    v16h fa0 = ldgc<K>(A, ao0 + k0);
    v16h fa1 = ldgc<K>(A, ao0 + 16 * K + k0);
    v16h fb0 = ldgc<K>(WT, bo0 + k0);
    v16h fb1 = ldgc<K>(WT, bo0 + 16 * K + k0);
    v16h fb2 = ldgc<K>(WT, bo0 + 32 * K + k0);
    v16h fb3 = ldgc<K>(WT, bo0 + 48 * K + k0);
    c[0] = wmma_f16(fa0, fb0, c[0]);
    c[1] = wmma_f16(fa0, fb1, c[1]);
    c[2] = wmma_f16(fa0, fb2, c[2]);
    c[3] = wmma_f16(fa0, fb3, c[3]);
    c[4] = wmma_f16(fa1, fb0, c[4]);
    c[5] = wmma_f16(fa1, fb1, c[5]);
    c[6] = wmma_f16(fa1, fb2, c[6]);
    c[7] = wmma_f16(fa1, fb3, c[7]);
  }
}

template <int K>
__global__ __launch_bounds__(256, 1)
void gemm_f16_kernel(const _Float16* __restrict__ A,
                     const _Float16* __restrict__ WT,
                     const float* __restrict__ bias,
                     _Float16* __restrict__ out, int ldo) {
  int w = (int)(threadIdx.x >> 5);
  int m0 = (int)blockIdx.y * 64 + (w & 1) * 32;
  int n0 = (int)blockIdx.x * 256 + (w >> 1) * 64;
  v8f z = {0.f, 0.f, 0.f, 0.f, 0.f, 0.f, 0.f, 0.f};
  v8f c[8] = {z, z, z, z, z, z, z, z};
  gemm_core<K>(A, WT, m0, n0, c);
  int lane = (int)(threadIdx.x & 31), cr = lane & 15, rb = (lane >> 4) * 8;
#pragma unroll
  for (int i = 0; i < 2; ++i)
#pragma unroll
    for (int j = 0; j < 4; ++j) {
      v8f acc = c[i * 4 + j];
      int col = n0 + 16 * j + cr;
      float bv = bias ? bias[col] : 0.f;
#pragma unroll
      for (int r = 0; r < 8; ++r) {
        int row = m0 + 16 * i + rb + r;
        out[row * ldo + col] = (_Float16)(acc[r] + bv);
      }
    }
}

template <int K>
__global__ __launch_bounds__(256, 1)
void gemm_f32res_kernel(const _Float16* __restrict__ A,
                        const _Float16* __restrict__ WT,
                        const float* __restrict__ bias,
                        const float* __restrict__ resid,
                        float* __restrict__ out, int ldo) {
  int w = (int)(threadIdx.x >> 5);
  int m0 = (int)blockIdx.y * 64 + (w & 1) * 32;
  int n0 = (int)blockIdx.x * 256 + (w >> 1) * 64;
  v8f z = {0.f, 0.f, 0.f, 0.f, 0.f, 0.f, 0.f, 0.f};
  v8f c[8] = {z, z, z, z, z, z, z, z};
  gemm_core<K>(A, WT, m0, n0, c);
  int lane = (int)(threadIdx.x & 31), cr = lane & 15, rb = (lane >> 4) * 8;
#pragma unroll
  for (int i = 0; i < 2; ++i)
#pragma unroll
    for (int j = 0; j < 4; ++j) {
      v8f acc = c[i * 4 + j];
      int col = n0 + 16 * j + cr;
      float bv = bias[col];
#pragma unroll
      for (int r = 0; r < 8; ++r) {
        int row = m0 + 16 * i + rb + r;
        int off = row * ldo + col;
        out[off] = acc[r] + bv + resid[off];
      }
    }
}

// ---------------------------------------------------------------- attention
// q: (B,S,NH*DH) f16 ; k: (B,T,NH*DH) f16 ; vT: (B,NH,DH,T) f16 ; out like q.
// grid: (S/64, NH, B), block 128 (4 waves); wave owns 16 query rows, streams
// keys in chunks of 32 with online (flash) softmax. T is compile-time.
template <int T>
__global__ __launch_bounds__(128, 1)
void attn_kernel(const _Float16* __restrict__ qm,
                 const _Float16* __restrict__ km,
                 const _Float16* __restrict__ vT,
                 _Float16* __restrict__ om,
                 int S, float scale) {
  int b = (int)blockIdx.z, h = (int)blockIdx.y;
  int w = (int)(threadIdx.x >> 5);
  int lane = (int)(threadIdx.x & 31), cr = lane & 15, rb = (lane >> 4) * 8;
  int i0 = ((int)blockIdx.x * 4 + w) * 16;

  int qoff = (b * S + i0) * DAc + h * DH;
  v16h qa0 = ldgc<DAc>(qm, qoff);
  v16h qa1 = ldgc<DAc>(qm, qoff + 32);

  v8f z = {0.f, 0.f, 0.f, 0.f, 0.f, 0.f, 0.f, 0.f};
  v8f o0 = z, o1 = z, o2 = z, o3 = z;
  float m[8], l[8];
#pragma unroll
  for (int r = 0; r < 8; ++r) { m[r] = -1e30f; l[r] = 0.f; }

  __shared__ __align__(16) _Float16 lp[4][16 * 32];
  _Float16* myp = lp[w];

  int kbase = b * T * DAc + h * DH;
  int vbase = ((b * NH + h) * DH) * T;

  for (int j0 = 0; j0 < T; j0 += 32) {
    int koff = kbase + j0 * DAc;
    v8f s0 = z, s1 = z;
    {
      v16h kb00 = ldgc<DAc>(km, koff);
      v16h kb10 = ldgc<DAc>(km, koff + 16 * DAc);
      s0 = wmma_f16(qa0, kb00, s0);
      s1 = wmma_f16(qa0, kb10, s1);
      v16h kb01 = ldgc<DAc>(km, koff + 32);
      v16h kb11 = ldgc<DAc>(km, koff + 16 * DAc + 32);
      s0 = wmma_f16(qa1, kb01, s0);
      s1 = wmma_f16(qa1, kb11, s1);
    }
    float tm[8];
#pragma unroll
    for (int r = 0; r < 8; ++r) {
      float a = s0[r] * scale, c2 = s1[r] * scale;
      s0[r] = a; s1[r] = c2;
      tm[r] = fmaxf(a, c2);
    }
#pragma unroll
    for (int r = 0; r < 8; ++r) {
      tm[r] = fmaxf(tm[r], __shfl_xor(tm[r], 1, 32));
      tm[r] = fmaxf(tm[r], __shfl_xor(tm[r], 2, 32));
      tm[r] = fmaxf(tm[r], __shfl_xor(tm[r], 4, 32));
      tm[r] = fmaxf(tm[r], __shfl_xor(tm[r], 8, 32));
    }
    float corr[8], ts[8];
#pragma unroll
    for (int r = 0; r < 8; ++r) {
      float mn = fmaxf(m[r], tm[r]);
      corr[r] = __expf(m[r] - mn);
      m[r] = mn;
      s0[r] = __expf(s0[r] - mn);
      s1[r] = __expf(s1[r] - mn);
      ts[r] = s0[r] + s1[r];
    }
#pragma unroll
    for (int r = 0; r < 8; ++r) {
      ts[r] += __shfl_xor(ts[r], 1, 32);
      ts[r] += __shfl_xor(ts[r], 2, 32);
      ts[r] += __shfl_xor(ts[r], 4, 32);
      ts[r] += __shfl_xor(ts[r], 8, 32);
      l[r] = l[r] * corr[r] + ts[r];
    }
#pragma unroll
    for (int r = 0; r < 8; ++r) {
      o0[r] *= corr[r]; o1[r] *= corr[r];
      o2[r] *= corr[r]; o3[r] *= corr[r];
    }
    // P (C-frag layout) -> LDS row-major [16][32] -> reload as A-frag.
#pragma unroll
    for (int r = 0; r < 8; ++r) {
      myp[(rb + r) * 32 + cr]      = (_Float16)s0[r];
      myp[(rb + r) * 32 + 16 + cr] = (_Float16)s1[r];
    }
    v16h pa = ldlds(myp);
    int voff = vbase + j0;
    o0 = wmma_f16(pa, ldgc<T>(vT, voff), o0);
    o1 = wmma_f16(pa, ldgc<T>(vT, voff + 16 * T), o1);
    o2 = wmma_f16(pa, ldgc<T>(vT, voff + 32 * T), o2);
    o3 = wmma_f16(pa, ldgc<T>(vT, voff + 48 * T), o3);
  }

  int obase = (b * S + i0) * DAc + h * DH;
#pragma unroll
  for (int r = 0; r < 8; ++r) {
    float inv = 1.0f / l[r];
    int roff = obase + (rb + r) * DAc;
    om[roff + cr]      = (_Float16)(o0[r] * inv);
    om[roff + 16 + cr] = (_Float16)(o1[r] * inv);
    om[roff + 32 + cr] = (_Float16)(o2[r] * inv);
    om[roff + 48 + cr] = (_Float16)(o3[r] * inv);
  }
}

// ---------------------------------------------------------------- LayerNorm
__global__ void ln_kernel(const float* __restrict__ x,
                          const float* __restrict__ g,
                          const float* __restrict__ bb,
                          _Float16* __restrict__ out) {
  const int D = 1024;
  int row = blockIdx.x, tid = threadIdx.x;
  const float* xr = x + (size_t)row * D;
  float v[4], s = 0.f, sq = 0.f;
#pragma unroll
  for (int i = 0; i < 4; ++i) {
    float t = xr[tid + i * 256];
    v[i] = t; s += t; sq += t * t;
  }
  __shared__ float red[256];
  red[tid] = s; __syncthreads();
  for (int st = 128; st > 0; st >>= 1) {
    if (tid < st) red[tid] += red[tid + st];
    __syncthreads();
  }
  float mean = red[0] / D;
  __syncthreads();
  red[tid] = sq; __syncthreads();
  for (int st = 128; st > 0; st >>= 1) {
    if (tid < st) red[tid] += red[tid + st];
    __syncthreads();
  }
  float var = red[0] / D - mean * mean;
  float rstd = rsqrtf(var + 1e-5f);
  _Float16* orow = out + (size_t)row * D;
#pragma unroll
  for (int i = 0; i < 4; ++i) {
    int c = tid + i * 256;
    orow[c] = (_Float16)((v[i] - mean) * rstd * g[c] + bb[c]);
  }
}

// ---------------------------------------------------------------- small utils
// w[K][N] f32  ->  wT[N][K] f16
__global__ void wtrans_kernel(const float* __restrict__ w,
                              _Float16* __restrict__ wT, int K, int N) {
  size_t idx = (size_t)blockIdx.x * 256 + threadIdx.x;
  if (idx >= (size_t)K * N) return;
  size_t n = idx / (size_t)K, k = idx % (size_t)K;
  wT[idx] = (_Float16)w[k * (size_t)N + n];
}

__global__ void tohalf_kernel(const float* __restrict__ in,
                              _Float16* __restrict__ out, size_t n) {
  size_t idx = (size_t)blockIdx.x * 256 + threadIdx.x;
  if (idx < n) out[idx] = (_Float16)in[idx];
}

// v (B,T,NH*DH) f16 -> vT (B,NH,DH,T) f16
__global__ void vtrans_kernel(const _Float16* __restrict__ v,
                              _Float16* __restrict__ vT, int T) {
  size_t idx = (size_t)blockIdx.x * 256 + threadIdx.x;
  size_t total = (size_t)Bc * NH * DH * (size_t)T;
  if (idx >= total) return;
  int t = (int)(idx % (size_t)T);
  size_t r = idx / (size_t)T;
  int d = (int)(r % DH); r /= DH;
  int h = (int)(r % NH);
  int b = (int)(r / NH);
  vT[idx] = v[((size_t)b * T + t) * (NH * DH) + h * DH + d];
}

// p (rows, 2*DFF) f16 -> a*gelu(gate) (rows, DFF) f16   (exact erf gelu)
__global__ void geglu_kernel(const _Float16* __restrict__ p,
                             _Float16* __restrict__ a, size_t rows) {
  size_t idx = (size_t)blockIdx.x * 256 + threadIdx.x;
  if (idx >= rows * (size_t)DFFc) return;
  size_t row = idx / (size_t)DFFc;
  int c = (int)(idx % (size_t)DFFc);
  float av = (float)p[row * (size_t)(2 * DFFc) + c];
  float g  = (float)p[row * (size_t)(2 * DFFc) + DFFc + c];
  float gl = 0.5f * g * (1.0f + erff(g * 0.70710678118654752f));
  a[idx] = (_Float16)(av * gl);
}

// ---------------------------------------------------------------- launcher
extern "C" void kernel_launch(void* const* d_in, const int* in_sizes, int n_in,
                              void* d_out, int out_size, void* d_ws, size_t ws_size,
                              hipStream_t stream) {
  (void)in_sizes; (void)n_in; (void)out_size; (void)ws_size;
  const float* x     = (const float*)d_in[0];
  const float* cond  = (const float*)d_in[1];
  const float* ln1g  = (const float*)d_in[2];
  const float* ln1b  = (const float*)d_in[3];
  const float* ln2g  = (const float*)d_in[4];
  const float* ln2b  = (const float*)d_in[5];
  const float* ln3g  = (const float*)d_in[6];
  const float* ln3b  = (const float*)d_in[7];
  const float* sa_wq = (const float*)d_in[8];
  const float* sa_wk = (const float*)d_in[9];
  const float* sa_wv = (const float*)d_in[10];
  const float* sa_wo = (const float*)d_in[11];
  const float* sa_bo = (const float*)d_in[12];
  const float* ca_wq = (const float*)d_in[13];
  const float* ca_wk = (const float*)d_in[14];
  const float* ca_wv = (const float*)d_in[15];
  const float* ca_wo = (const float*)d_in[16];
  const float* ca_bo = (const float*)d_in[17];
  const float* ff_w1 = (const float*)d_in[18];
  const float* ff_b1 = (const float*)d_in[19];
  const float* ff_w2 = (const float*)d_in[20];
  const float* ff_b2 = (const float*)d_in[21];
  float* outp = (float*)d_out;

  const int B = Bc, S = Sc, T = Tcc, DM = DMc, DC = DCc, DA = DAc, DFF = DFFc;
  const size_t rowsA = (size_t)B * S;           // 8192
  const size_t rowsC = (size_t)B * T;           // 2048

  char* wp = (char*)d_ws;
  auto alloc = [&](size_t bytes) -> char* {
    char* r = wp; wp += (bytes + 255) & ~(size_t)255; return r;
  };
  float*    xres  = (float*)   alloc(rowsA * DM * 4);
  _Float16* hf16  = (_Float16*)alloc(rowsA * DM * 2);
  _Float16* qf16  = (_Float16*)alloc(rowsA * DA * 2);
  _Float16* kf16  = (_Float16*)alloc(rowsA * DA * 2);
  _Float16* vf16  = (_Float16*)alloc(rowsA * DA * 2);
  _Float16* vtb   = (_Float16*)alloc(rowsA * DA * 2);
  _Float16* of16  = (_Float16*)alloc(rowsA * DA * 2);
  _Float16* cf16  = (_Float16*)alloc(rowsC * DC * 2);
  _Float16* sa_wqT = (_Float16*)alloc((size_t)DM * DA * 2);
  _Float16* sa_wkT = (_Float16*)alloc((size_t)DM * DA * 2);
  _Float16* sa_wvT = (_Float16*)alloc((size_t)DM * DA * 2);
  _Float16* sa_woT = (_Float16*)alloc((size_t)DA * DM * 2);
  _Float16* ca_wqT = (_Float16*)alloc((size_t)DM * DA * 2);
  _Float16* ca_wkT = (_Float16*)alloc((size_t)DC * DA * 2);
  _Float16* ca_wvT = (_Float16*)alloc((size_t)DC * DA * 2);
  _Float16* ca_woT = (_Float16*)alloc((size_t)DA * DM * 2);
  _Float16* w1T    = (_Float16*)alloc((size_t)DM * 2 * DFF * 2);
  _Float16* w2T    = (_Float16*)alloc((size_t)DFF * DM * 2);
  _Float16* pf16   = (_Float16*)alloc(rowsA * 2 * DFF * 2);
  _Float16* af16   = (_Float16*)alloc(rowsA * DFF * 2);

  auto nb = [](size_t n) { return (unsigned)((n + 255) / 256); };

  // residual stream init
  hipMemcpyAsync(xres, x, rowsA * DM * 4, hipMemcpyDeviceToDevice, stream);

  // weight conversion + transpose (f32 KxN -> f16 NxK)
  wtrans_kernel<<<nb((size_t)DM * DA), 256, 0, stream>>>(sa_wq, sa_wqT, DM, DA);
  wtrans_kernel<<<nb((size_t)DM * DA), 256, 0, stream>>>(sa_wk, sa_wkT, DM, DA);
  wtrans_kernel<<<nb((size_t)DM * DA), 256, 0, stream>>>(sa_wv, sa_wvT, DM, DA);
  wtrans_kernel<<<nb((size_t)DA * DM), 256, 0, stream>>>(sa_wo, sa_woT, DA, DM);
  wtrans_kernel<<<nb((size_t)DM * DA), 256, 0, stream>>>(ca_wq, ca_wqT, DM, DA);
  wtrans_kernel<<<nb((size_t)DC * DA), 256, 0, stream>>>(ca_wk, ca_wkT, DC, DA);
  wtrans_kernel<<<nb((size_t)DC * DA), 256, 0, stream>>>(ca_wv, ca_wvT, DC, DA);
  wtrans_kernel<<<nb((size_t)DA * DM), 256, 0, stream>>>(ca_wo, ca_woT, DA, DM);
  wtrans_kernel<<<nb((size_t)DM * 2 * DFF), 256, 0, stream>>>(ff_w1, w1T, DM, 2 * DFF);
  wtrans_kernel<<<nb((size_t)DFF * DM), 256, 0, stream>>>(ff_w2, w2T, DFF, DM);
  tohalf_kernel<<<nb(rowsC * DC), 256, 0, stream>>>(cond, cf16, rowsC * DC);

  const float scale = 0.125f; // 1/sqrt(64)
  dim3 gA(DA / 256, (unsigned)(rowsA / 64));         // N=1024, M=8192
  dim3 gC(DA / 256, (unsigned)(rowsC / 64));         // N=1024, M=2048
  dim3 gF1((2 * DFF) / 256, (unsigned)(rowsA / 64)); // N=8192, M=8192
  dim3 attnSA(S / 64, NH, B);

  // ---- self attention ----
  ln_kernel<<<(unsigned)rowsA, 256, 0, stream>>>(xres, ln1g, ln1b, hf16);
  gemm_f16_kernel<DMc><<<gA, 256, 0, stream>>>(hf16, sa_wqT, nullptr, qf16, DA);
  gemm_f16_kernel<DMc><<<gA, 256, 0, stream>>>(hf16, sa_wkT, nullptr, kf16, DA);
  gemm_f16_kernel<DMc><<<gA, 256, 0, stream>>>(hf16, sa_wvT, nullptr, vf16, DA);
  vtrans_kernel<<<nb(rowsA * DA), 256, 0, stream>>>(vf16, vtb, S);
  attn_kernel<Sc><<<attnSA, 128, 0, stream>>>(qf16, kf16, vtb, of16, S, scale);
  gemm_f32res_kernel<DAc><<<gA, 256, 0, stream>>>(of16, sa_woT, sa_bo, xres, xres, DM);

  // ---- cross attention ----
  ln_kernel<<<(unsigned)rowsA, 256, 0, stream>>>(xres, ln2g, ln2b, hf16);
  gemm_f16_kernel<DMc><<<gA, 256, 0, stream>>>(hf16, ca_wqT, nullptr, qf16, DA);
  gemm_f16_kernel<DCc><<<gC, 256, 0, stream>>>(cf16, ca_wkT, nullptr, kf16, DA);
  gemm_f16_kernel<DCc><<<gC, 256, 0, stream>>>(cf16, ca_wvT, nullptr, vf16, DA);
  vtrans_kernel<<<nb((size_t)B * NH * DH * T), 256, 0, stream>>>(vf16, vtb, T);
  attn_kernel<Tcc><<<attnSA, 128, 0, stream>>>(qf16, kf16, vtb, of16, S, scale);
  gemm_f32res_kernel<DAc><<<gA, 256, 0, stream>>>(of16, ca_woT, ca_bo, xres, xres, DM);

  // ---- GEGLU feed-forward ----
  ln_kernel<<<(unsigned)rowsA, 256, 0, stream>>>(xres, ln3g, ln3b, hf16);
  gemm_f16_kernel<DMc><<<gF1, 256, 0, stream>>>(hf16, w1T, ff_b1, pf16, 2 * DFF);
  geglu_kernel<<<nb(rowsA * DFF), 256, 0, stream>>>(pf16, af16, rowsA);
  gemm_f32res_kernel<DFFc><<<gA, 256, 0, stream>>>(af16, w2T, ff_b2, xres, outp, DM);
}
